// WGCN_85890755986036
// MI455X (gfx1250) — matrix operand order repaired and checked by reading, not compiled, not run on hardware
//
#include <hip/hip_runtime.h>

// ---------------------------------------------------------------------------
// WGCN: out = relu((A .* E) @ X @ W + b), N=12000, C=F=128
// Memory-bound on streaming A,E (1.15 GB @ 23.3 TB/s => ~49us floor).
// bf16 WMMA w/ fp32 accumulate; B tiles staged to LDS via async-to-LDS.
// ---------------------------------------------------------------------------

typedef __attribute__((ext_vector_type(16))) __bf16 v16bf;
typedef __attribute__((ext_vector_type(8)))  __bf16 v8bf;
typedef __attribute__((ext_vector_type(8)))  float  v8f;
typedef __attribute__((ext_vector_type(4)))  float  v4f;

typedef int i4 __attribute__((vector_size(16)));
typedef __attribute__((address_space(1))) i4* as1_i4p;   // global
typedef __attribute__((address_space(3))) i4* as3_i4p;   // LDS

#define N_NODES 12000
#define C_DIM   128
#define F_DIM   128
#define KSPLIT  15
#define KSEG    (N_NODES / KSPLIT)          // 800
#define CHUNKS  (KSEG / 32)                 // 25 K-chunks of 32 per segment
#define ROW_BLOCKS ((N_NODES + 127) / 128)  // 94
#define B_ROW   40                          // LDS row stride (80B): conflict-free b128

// ws layout (bytes)
#define XT_OFF  0
#define XT_BYTES (C_DIM * N_NODES * 2)          // 3,072,000
#define WT_OFF  (XT_OFF + XT_BYTES)
#define WT_BYTES (C_DIM * F_DIM * 2)            // 32,768
#define AGG_OFF (WT_OFF + WT_BYTES)             // 3,104,768 (16B aligned)

#if __has_builtin(__builtin_amdgcn_global_load_async_to_lds_b128)
#define HAS_ASYNC_LDS 1
#else
#define HAS_ASYNC_LDS 0
#endif

static __device__ __forceinline__ __bf16 f2bf(float f) { return (__bf16)f; }

#if HAS_ASYNC_LDS
static __device__ __forceinline__ void async_copy16(const void* g, void* l) {
  __builtin_amdgcn_global_load_async_to_lds_b128(
      (as1_i4p)(void*)g, (as3_i4p)l, 0, 0);
}
#endif

// ---------------------------------------------------------------------------
// Prep: XT[n][k] = bf16(x[k][n]); WT[n][k] = bf16(W[k][n]); agg = 0
// ---------------------------------------------------------------------------
__global__ void __launch_bounds__(256) wgcn_prep(
    const float* __restrict__ x, const float* __restrict__ W,
    __bf16* __restrict__ XT, __bf16* __restrict__ WT,
    float* __restrict__ agg) {
  const int tid    = blockIdx.x * blockDim.x + threadIdx.x;
  const int stride = gridDim.x * blockDim.x;
  for (int i = tid; i < N_NODES * C_DIM; i += stride) {
    const int k = i / C_DIM;
    const int n = i % C_DIM;
    XT[(size_t)n * N_NODES + k] = f2bf(x[i]);
    agg[i] = 0.0f;
  }
  for (int i = tid; i < C_DIM * F_DIM; i += stride) {
    const int k = i / F_DIM;
    const int n = i % F_DIM;
    WT[(size_t)n * C_DIM + k] = f2bf(W[i]);
  }
}

// ---------------------------------------------------------------------------
// Stage 1: agg += (A .* E) @ X   over one K segment per block-row.
// 8 waves/block, each wave: 16 output rows x 128 cols (8 WMMA accumulators).
// B tile (32k x 128n, 8KB) is staged block-cooperatively into LDS (async,
// double-buffered); A.E stream is read once with non-temporal b128 loads.
// ---------------------------------------------------------------------------
__global__ void __launch_bounds__(256) wgcn_spmm(
    const float* __restrict__ A, const float* __restrict__ E,
    const __bf16* __restrict__ XT, float* __restrict__ agg) {
  __shared__ __bf16 Bbuf[2][C_DIM * B_ROW];   // 2 x 10,240 B

  const int tid  = (int)threadIdx.x;
  const int wave = tid >> 5;
  const int lane = tid & 31;
  const int r0   = (int)blockIdx.x * 128 + wave * 16;
  const int ln   = lane & 15;
  const int hi   = lane >> 4;
  const bool active = (r0 < N_NODES);
  const int rowA = (r0 + ln < N_NODES) ? (r0 + ln) : (N_NODES - 1);  // clamp idle tail

  const float* __restrict__ arow = A + (size_t)rowA * N_NODES;
  const float* __restrict__ erow = E + (size_t)rowA * N_NODES;
  const int kseg0 = (int)blockIdx.y * KSEG;

  // Block-cooperative stage of one 32x128 XT chunk into LDS (512 x 16B).
  auto stage = [&](int c0, int bufsel) {
#pragma unroll
    for (int s = tid; s < 512; s += 256) {
      const int n  = s >> 2;
      const int ko = (s & 3) * 8;
      const __bf16* g = XT + (size_t)n * N_NODES + c0 + ko;
      __bf16* l = &Bbuf[bufsel][n * B_ROW + ko];
#if HAS_ASYNC_LDS
      async_copy16(g, l);
#else
      *(v8bf*)l = *(const v8bf*)g;
#endif
    }
  };

  v8f c[8];
#pragma unroll
  for (int t = 0; t < 8; ++t)
#pragma unroll
    for (int j = 0; j < 8; ++j) c[t][j] = 0.0f;

  stage(kseg0, 0);

  for (int ch = 0; ch < CHUNKS; ++ch) {
    const int c0 = kseg0 + ch * 32;
#if HAS_ASYNC_LDS
    asm volatile("s_wait_asynccnt 0x0" ::: "memory");  // own async stage done
#endif
    __syncthreads();                                   // publish buf[ch&1]
    if (ch + 1 < CHUNKS) stage(c0 + 32, (ch + 1) & 1); // overlaps compute below

    // A-fragment 16x32 bf16 of (A .* E), ISA layout:
    //  lane<16: V0-3=K 0..7, V4-7=K 16..23 | lane>=16: V0-3=K 8..15, V4-7=K 24..31
    // Non-temporal: 1.15GB stream read once, keep XT resident in L2.
    const int ka0 = c0 + hi * 8;
    const int ka1 = ka0 + 16;
    v4f a0 = __builtin_nontemporal_load((const v4f*)(arow + ka0));
    v4f a1 = __builtin_nontemporal_load((const v4f*)(arow + ka0 + 4));
    v4f a2 = __builtin_nontemporal_load((const v4f*)(arow + ka1));
    v4f a3 = __builtin_nontemporal_load((const v4f*)(arow + ka1 + 4));
    v4f e0 = __builtin_nontemporal_load((const v4f*)(erow + ka0));
    v4f e1 = __builtin_nontemporal_load((const v4f*)(erow + ka0 + 4));
    v4f e2 = __builtin_nontemporal_load((const v4f*)(erow + ka1));
    v4f e3 = __builtin_nontemporal_load((const v4f*)(erow + ka1 + 4));
    v16bf af;
#pragma unroll
    for (int j = 0; j < 4; ++j) {
      af[j]      = f2bf(a0[j] * e0[j]);
      af[4 + j]  = f2bf(a1[j] * e1[j]);
      af[8 + j]  = f2bf(a2[j] * e2[j]);
      af[12 + j] = f2bf(a3[j] * e3[j]);
    }

    // All 8 B-fragments from LDS first (one wait), then 8 WMMAs back-to-back.
    const __bf16* bb = &Bbuf[ch & 1][hi * 8];
    v16bf bm[8];
#pragma unroll
    for (int t = 0; t < 8; ++t) {
      const __bf16* bp = bb + (t * 16 + ln) * B_ROW;
      v8bf b0 = *(const v8bf*)bp;
      v8bf b1 = *(const v8bf*)(bp + 16);
#pragma unroll
      for (int j = 0; j < 8; ++j) { bm[t][j] = b0[j]; bm[t][8 + j] = b1[j]; }
    }
#pragma unroll
    for (int t = 0; t < 8; ++t)
      c[t] = __builtin_amdgcn_wmma_f32_16x16x32_bf16(false, af, false, bm[t],
                                                     (short)0, c[t], false, false);
  }

  // C/D layout: lane<16 -> (M=v, N=ln); lane>=16 -> (M=v+8, N=ln)
  if (active) {
#pragma unroll
    for (int t = 0; t < 8; ++t) {
      const int n = t * 16 + ln;
#pragma unroll
      for (int v = 0; v < 8; ++v) {
        const int rr = r0 + v + hi * 8;
        atomicAdd(agg + (size_t)rr * C_DIM + n, c[t][v]);
      }
    }
  }
}

// ---------------------------------------------------------------------------
// Stage 2: out = relu(agg @ W + b), K = 128 (4 chunks of 32)
// ---------------------------------------------------------------------------
__global__ void __launch_bounds__(256) wgcn_proj(
    const float* __restrict__ agg, const __bf16* __restrict__ WT,
    const float* __restrict__ bias, float* __restrict__ out) {
  const int wave = (int)(threadIdx.x >> 5);
  const int lane = (int)(threadIdx.x & 31);
  const int r0   = (int)blockIdx.x * 128 + wave * 16;
  if (r0 >= N_NODES) return;   // no barriers in this kernel
  const int ln = lane & 15;
  const int hi = lane >> 4;

  const float* __restrict__ grow = agg + (size_t)(r0 + ln) * C_DIM;

  v8f c[8];
#pragma unroll
  for (int t = 0; t < 8; ++t)
#pragma unroll
    for (int j = 0; j < 8; ++j) c[t][j] = 0.0f;

#pragma unroll
  for (int ch = 0; ch < 4; ++ch) {
    const int c0  = ch * 32;
    const int ka0 = c0 + hi * 8;
    const int ka1 = ka0 + 16;
    v4f a0 = *(const v4f*)(grow + ka0);
    v4f a1 = *(const v4f*)(grow + ka0 + 4);
    v4f a2 = *(const v4f*)(grow + ka1);
    v4f a3 = *(const v4f*)(grow + ka1 + 4);
    v16bf af;
#pragma unroll
    for (int j = 0; j < 4; ++j) {
      af[j]      = f2bf(a0[j]);
      af[4 + j]  = f2bf(a1[j]);
      af[8 + j]  = f2bf(a2[j]);
      af[12 + j] = f2bf(a3[j]);
    }
#pragma unroll
    for (int t = 0; t < 8; ++t) {
      const __bf16* bp = WT + (size_t)(t * 16 + ln) * C_DIM + c0 + hi * 8;
      v8bf b0 = *(const v8bf*)bp;
      v8bf b1 = *(const v8bf*)(bp + 16);
      v16bf bm;
#pragma unroll
      for (int j = 0; j < 8; ++j) { bm[j] = b0[j]; bm[8 + j] = b1[j]; }
      c[t] = __builtin_amdgcn_wmma_f32_16x16x32_bf16(false, af, false, bm,
                                                     (short)0, c[t], false, false);
    }
  }
#pragma unroll
  for (int t = 0; t < 8; ++t) {
    const int n  = t * 16 + ln;
    const float bv = bias[n];
#pragma unroll
    for (int v = 0; v < 8; ++v) {
      const int rr = r0 + v + hi * 8;
      const float o = c[t][v] + bv;
      out[(size_t)rr * F_DIM + n] = o > 0.0f ? o : 0.0f;
    }
  }
}

// ---------------------------------------------------------------------------
extern "C" void kernel_launch(void* const* d_in, const int* in_sizes, int n_in,
                              void* d_out, int out_size, void* d_ws, size_t ws_size,
                              hipStream_t stream) {
  (void)in_sizes; (void)n_in; (void)out_size; (void)ws_size;
  const float* A = (const float*)d_in[0];   // adjacency [N,N]
  const float* E = (const float*)d_in[1];   // edge_weights [N,N]
  const float* x = (const float*)d_in[2];   // [N,C]
  const float* W = (const float*)d_in[3];   // [C,F]
  const float* b = (const float*)d_in[4];   // [F]
  float* out = (float*)d_out;               // [N,F]

  char* ws = (char*)d_ws;
  __bf16* XT = (__bf16*)(ws + XT_OFF);
  __bf16* WT = (__bf16*)(ws + WT_OFF);
  float*  agg = (float*)(ws + AGG_OFF);

  wgcn_prep<<<1024, 256, 0, stream>>>(x, W, XT, WT, agg);
  dim3 grid_spmm(ROW_BLOCKS, KSPLIT);
  wgcn_spmm<<<grid_spmm, 256, 0, stream>>>(A, E, XT, agg);
  wgcn_proj<<<ROW_BLOCKS, 256, 0, stream>>>(agg, WT, b, out);
}
